// TwoLayerGCN_2035814498363
// MI455X (gfx1250) — matrix operand order repaired and checked by reading, not compiled, not run on hardware
//
#include <hip/hip_runtime.h>
#include <hip/hip_bf16.h>

typedef __attribute__((ext_vector_type(16))) _Float16 v16h;
typedef __attribute__((ext_vector_type(8)))  float    v8f;

#define TPB 256
#define GEMM_WAVES_M 512   // waves striding over M-tiles per output-column tile

// ---------------- degree / normalization ----------------

__global__ void k_deg_init(float* __restrict__ deg, int N) {
  int i = blockIdx.x * blockDim.x + threadIdx.x;
  if (i < N) deg[i] = 1.0f;  // self-loop weight 1 at every node
}

__global__ void k_deg_accum(const int* __restrict__ col, const float* __restrict__ w,
                            float* __restrict__ deg, int E) {
  int e = blockIdx.x * blockDim.x + threadIdx.x;
  if (e < E) unsafeAtomicAdd(&deg[col[e]], w[e]);   // global_atomic_add_f32, resolves in L2
}

__global__ void k_deg_to_dis(float* __restrict__ deg, int N) {
  int i = blockIdx.x * blockDim.x + threadIdx.x;
  if (i < N) {
    float d = deg[i];
    deg[i] = d > 0.0f ? rsqrtf(d) : 0.0f;
  }
}

// ---------------- embedding gather ----------------

__global__ void k_embed(const int* __restrict__ x, const float* __restrict__ emb,
                        float* __restrict__ h, int N, int D) {
  int idx = blockIdx.x * blockDim.x + threadIdx.x;
  int per = D >> 2;                 // float4 per row
  int node = idx / per;
  int d4 = (idx - node * per) << 2;
  if (node >= N) return;
  const float4 v = *reinterpret_cast<const float4*>(emb + (size_t)x[node] * D + d4);
  *reinterpret_cast<float4*>(h + (size_t)node * D + d4) = v;
}

// ---------------- WMMA GEMM: C[M,Nout] = A[M,K] * B[K,Nout], K = KCH*32 ----------------
// Wave-per-column-tile with register-resident B: each wave loads its 128x16 B slice
// once (4 x v16h fragments), then grid-strides over M-tiles doing 4 unrolled
// v_wmma_f32_16x16x32_f16 per tile. A is streamed with aligned float4 loads.
//
// Fragment layouts (ISA 7.12.2):
//   A 16x32 f16: lane L<16 -> row L, elems 0..7 = K{0..7}, 8..15 = K{16..23};
//                lane L>=16 -> row L-16, K{8..15}/{24..31}.
//   B 32x16 f16: lanes 0..15 -> col=lane, elems = K{0..7},{16..23}; lanes 16..31 -> col=lane-16, K{8..15}/{24..31}.
//   C/D 16x16 f32: lanes 0..15 -> col=lane rows 0..7; lanes 16..31 -> col=lane-16 rows 8..15.

template <int KCH>
__global__ void k_gemm_wmma(const float* __restrict__ A, const float* __restrict__ B,
                            float* __restrict__ C, int M, int Nout) {
  const int K = KCH * 32;
  int gwave = (int)((blockIdx.x * blockDim.x + threadIdx.x) >> 5);
  int lane  = threadIdx.x & 31;
  int tileN = gwave / GEMM_WAVES_M;   // launcher sizes grid exactly: tilesN*GEMM_WAVES_M waves
  int waveM = gwave - tileN * GEMM_WAVES_M;

  int lmod = lane & 15;
  int lhi  = lane >> 4;               // 0 or 1
  int bcol = tileN * 16 + lmod;

  // ---- preload B fragments for this column tile (once per wave) ----
  v16h breg[KCH];
#pragma unroll
  for (int kc = 0; kc < KCH; ++kc) {
    int kb = kc * 32 + lhi * 8;
#pragma unroll
    for (int i = 0; i < 16; ++i)
      breg[kc][i] = (_Float16)B[(size_t)(kb + i + (i >= 8 ? 8 : 0)) * Nout + bcol];
  }

  // ---- stream M tiles ----
  for (int tileM = waveM; tileM * 16 < M; tileM += GEMM_WAVES_M) {
    const float* Arow = A + (size_t)(tileM * 16 + lmod) * K;
    v8f acc = {};
#pragma unroll
    for (int kc = 0; kc < KCH; ++kc) {
      int ka = kc * 32 + lhi * 8;     // multiple of 8 -> float4 aligned
      float4 a0 = *reinterpret_cast<const float4*>(Arow + ka);
      float4 a1 = *reinterpret_cast<const float4*>(Arow + ka + 4);
      float4 a2 = *reinterpret_cast<const float4*>(Arow + ka + 16);
      float4 a3 = *reinterpret_cast<const float4*>(Arow + ka + 20);
      v16h a;
      a[0] = (_Float16)a0.x; a[1] = (_Float16)a0.y; a[2] = (_Float16)a0.z; a[3] = (_Float16)a0.w;
      a[4] = (_Float16)a1.x; a[5] = (_Float16)a1.y; a[6] = (_Float16)a1.z; a[7] = (_Float16)a1.w;
      a[8]  = (_Float16)a2.x; a[9]  = (_Float16)a2.y; a[10] = (_Float16)a2.z; a[11] = (_Float16)a2.w;
      a[12] = (_Float16)a3.x; a[13] = (_Float16)a3.y; a[14] = (_Float16)a3.z; a[15] = (_Float16)a3.w;
      acc = __builtin_amdgcn_wmma_f32_16x16x32_f16(
          /*neg_a=*/false, a, /*neg_b=*/false, breg[kc],
          /*c_mod=*/(short)0, acc, /*reuse_a=*/false, /*reuse_b=*/false);
    }
    int m0 = tileM * 16 + lhi * 8;
#pragma unroll
    for (int r = 0; r < 8; ++r)
      C[(size_t)(m0 + r) * Nout + bcol] = acc[r];
  }
}

// ---------------- aggregation: dst = b + self + scatter(edges) ----------------

__global__ void k_agg_init(const float* __restrict__ src, const float* __restrict__ bias,
                           const float* __restrict__ dis, float* __restrict__ dst,
                           int N, int D) {
  int idx = blockIdx.x * blockDim.x + threadIdx.x;
  if (idx >= N * D) return;
  int node = idx / D;
  int d = idx - node * D;
  float s = dis[node];                       // self-loop norm = dis^2 (w=1)
  dst[idx] = bias[d] + src[idx] * s * s;
}

__global__ void k_agg_edges(const float* __restrict__ src, float* __restrict__ dst,
                            const int* __restrict__ row, const int* __restrict__ col,
                            const float* __restrict__ w, const float* __restrict__ dis,
                            int E, int D) {
  int idx = blockIdx.x * blockDim.x + threadIdx.x;
  int per = D >> 2;                          // 4 dims per thread
  int e = idx / per;
  if (e >= E) return;
  int d4 = (idx - e * per) << 2;
  int r = row[e], c = col[e];
  float norm = dis[r] * w[e] * dis[c];
  const float* sp = src + (size_t)r * D + d4;
  float* dp = dst + (size_t)c * D + d4;
  unsafeAtomicAdd(dp + 0, sp[0] * norm);
  unsafeAtomicAdd(dp + 1, sp[1] * norm);
  unsafeAtomicAdd(dp + 2, sp[2] * norm);
  unsafeAtomicAdd(dp + 3, sp[3] * norm);
}

__global__ void k_relu(float* __restrict__ h, int n) {
  int idx = blockIdx.x * blockDim.x + threadIdx.x;
  if (idx < n) h[idx] = fmaxf(h[idx], 0.0f);
}

// ---------------- log_softmax over 32 classes: one wave32 per row ----------------

__global__ void k_log_softmax32(float* __restrict__ out, int N) {
  int wave = (int)((blockIdx.x * blockDim.x + threadIdx.x) >> 5);
  int lane = threadIdx.x & 31;
  if (wave >= N) return;                     // wave-uniform exit
  float v = out[(size_t)wave * 32 + lane];
  float m = v;
#pragma unroll
  for (int off = 16; off; off >>= 1) m = fmaxf(m, __shfl_xor(m, off, 32));
  float e = expf(v - m);
  float s = e;
#pragma unroll
  for (int off = 16; off; off >>= 1) s += __shfl_xor(s, off, 32);
  out[(size_t)wave * 32 + lane] = v - m - logf(s);
}

// ---------------- launcher ----------------

static inline int cdiv(long long a, long long b) { return (int)((a + b - 1) / b); }

extern "C" void kernel_launch(void* const* d_in, const int* in_sizes, int n_in,
                              void* d_out, int out_size, void* d_ws, size_t ws_size,
                              hipStream_t stream) {
  const int*   x    = (const int*)d_in[0];
  const int*   ei   = (const int*)d_in[1];   // [2, E]
  const float* ew   = (const float*)d_in[2];
  const float* emb  = (const float*)d_in[3];
  const float* W1   = (const float*)d_in[4];
  const float* b1   = (const float*)d_in[5];
  const float* W2   = (const float*)d_in[6];
  const float* b2   = (const float*)d_in[7];
  float*       out  = (float*)d_out;

  const int N = in_sizes[0];
  const int E = in_sizes[2];
  const int D = in_sizes[5];                 // 128
  const int C = in_sizes[7];                 // 32
  const int* row = ei;
  const int* col = ei + E;

  float* dis  = (float*)d_ws;                // N
  float* bufA = dis + N;                     // N*D (embeddings -> h1)
  float* bufB = bufA + (size_t)N * D;        // N*D (xW products)

  // 1) symmetric normalization coefficients
  k_deg_init<<<cdiv(N, TPB), TPB, 0, stream>>>(dis, N);
  k_deg_accum<<<cdiv(E, TPB), TPB, 0, stream>>>(col, ew, dis, E);
  k_deg_to_dis<<<cdiv(N, TPB), TPB, 0, stream>>>(dis, N);

  // 2) embedding lookup -> bufA
  k_embed<<<cdiv((long long)N * (D / 4), TPB), TPB, 0, stream>>>(x, emb, bufA, N, D);

  // 3) layer 1: h W1 (WMMA) -> bufB, then normalized aggregation -> bufA, ReLU
  {
    long long waves = (long long)(D / 16) * GEMM_WAVES_M;   // tilesN * waves_m, exact grid
    k_gemm_wmma<4><<<cdiv(waves * 32, TPB), TPB, 0, stream>>>(bufA, W1, bufB, N, D);
  }
  k_agg_init<<<cdiv((long long)N * D, TPB), TPB, 0, stream>>>(bufB, b1, dis, bufA, N, D);
  k_agg_edges<<<cdiv((long long)E * (D / 4), TPB), TPB, 0, stream>>>(bufB, bufA, row, col,
                                                                    ew, dis, E, D);
  k_relu<<<cdiv((long long)N * D, TPB), TPB, 0, stream>>>(bufA, N * D);

  // 4) layer 2: h1 W2 (WMMA) -> bufB[:, :C], aggregation -> d_out
  {
    long long waves = (long long)(C / 16) * GEMM_WAVES_M;
    k_gemm_wmma<4><<<cdiv(waves * 32, TPB), TPB, 0, stream>>>(bufA, W2, bufB, N, C);
  }
  k_agg_init<<<cdiv((long long)N * C, TPB), TPB, 0, stream>>>(bufB, b2, dis, out, N, C);
  k_agg_edges<<<cdiv((long long)E * (C / 4), TPB), TPB, 0, stream>>>(bufB, out, row, col,
                                                                    ew, dis, E, C);

  // 5) row-wise log_softmax in place on d_out (one wave per row)
  k_log_softmax32<<<cdiv((long long)N * 32, TPB), TPB, 0, stream>>>(out, N);
}
